// AttentionPooling_2095944040957
// MI455X (gfx1250) — matrix operand order repaired and checked by reading, not compiled, not run on hardware
//
#include <hip/hip_runtime.h>
#include <hip/hip_bf16.h>

typedef __attribute__((ext_vector_type(16))) __bf16 v16bf;
typedef __attribute__((ext_vector_type(8)))  __bf16 v8bf;
typedef __attribute__((ext_vector_type(8)))  float  v8f;

#define HIDDEN 256
#define KPAD   264     // pad K dim (bf16 elems) to dodge LDS bank conflicts
#define CHUNK  512     // nodes per block in pass 2

// Branch-free tanh: native v_tanh_f32 on gfx1250 if clang exposes it,
// otherwise a 2-trans-op formula with no exec-mask branches.
__device__ __forceinline__ float fast_tanh(float v) {
#if __has_builtin(__builtin_amdgcn_tanhf)
  return __builtin_amdgcn_tanhf(v);
#else
  float t = __expf(-2.0f * __builtin_fabsf(v));        // underflows to 0, never overflows
  float r = (1.0f - t) * __builtin_amdgcn_rcpf(1.0f + t);
  return __builtin_copysignf(r, v);
#endif
}

// ---------------- Pass 1: e[n] = exp(tanh(x@W1+b1)@W2 + b2), denom[g] += e[n]
__global__ __launch_bounds__(256) void ap_pass1(
    const float* __restrict__ x, const int* __restrict__ batch,
    const float* __restrict__ W1, const float* __restrict__ b1,
    const float* __restrict__ W2, const float* __restrict__ b2p,
    float* __restrict__ eArr, float* __restrict__ denom, int nTiles)
{
  // W1 transposed [Nout][K] in bf16: B-fragment reads are two contiguous 8-K runs.
  __shared__ __bf16 sW1t[HIDDEN * KPAD];
  __shared__ float  sB1[HIDDEN];
  __shared__ float  sW2[HIDDEN];

  for (int idx = threadIdx.x; idx < HIDDEN * HIDDEN; idx += blockDim.x) {
    int K = idx >> 8;          // coalesced global read of W1[K][N]
    int Nc = idx & 255;
    sW1t[Nc * KPAD + K] = (__bf16)W1[idx];
  }
  if (threadIdx.x < HIDDEN) {
    sB1[threadIdx.x] = b1[threadIdx.x];
    sW2[threadIdx.x] = W2[threadIdx.x];
  }
  __syncthreads();

  const int lane   = threadIdx.x & 31;
  const int wave   = threadIdx.x >> 5;
  const int lane16 = lane & 15;
  const int hi     = lane >> 4;
  const float b2v  = b2p[0];

  for (int tile = blockIdx.x * 8 + wave; tile < nTiles; tile += gridDim.x * 8) {
    // ---- A fragments: 16 rows x 256 K of x, converted f32 -> bf16 ----
    const float* xrow = x + (size_t)(tile * 16 + lane16) * HIDDEN;
    v16bf a[8];
#pragma unroll
    for (int kt = 0; kt < 8; ++kt) {
      // lanes 0-15: elems 0..7 = K kt*32+0..7, elems 8..15 = K kt*32+16..23
      // lanes 16-31: same +8
      const float* p0 = xrow + kt * 32 + hi * 8;
      const float* p1 = p0 + 16;
      float4 f0 = *(const float4*)(p0);
      float4 f1 = *(const float4*)(p0 + 4);
      float4 f2 = *(const float4*)(p1);
      float4 f3 = *(const float4*)(p1 + 4);
      v16bf av;
      av[0]  = (__bf16)f0.x; av[1]  = (__bf16)f0.y; av[2]  = (__bf16)f0.z; av[3]  = (__bf16)f0.w;
      av[4]  = (__bf16)f1.x; av[5]  = (__bf16)f1.y; av[6]  = (__bf16)f1.z; av[7]  = (__bf16)f1.w;
      av[8]  = (__bf16)f2.x; av[9]  = (__bf16)f2.y; av[10] = (__bf16)f2.z; av[11] = (__bf16)f2.w;
      av[12] = (__bf16)f3.x; av[13] = (__bf16)f3.y; av[14] = (__bf16)f3.z; av[15] = (__bf16)f3.w;
      a[kt] = av;
    }

    float part[8] = {0.f, 0.f, 0.f, 0.f, 0.f, 0.f, 0.f, 0.f};

    for (int n = 0; n < 16; ++n) {          // 16 output-column blocks of 16
      v8f acc = {};
#pragma unroll
      for (int kt = 0; kt < 8; ++kt) {      // K = 256 in steps of 32
        // B fragment: lane holds col N = n*16+lane16, K = kt*32 + hi*8 .. +7 and kt*32 + 16 + hi*8 .. +7
        const __bf16* bp = &sW1t[(n * 16 + lane16) * KPAD + kt * 32 + hi * 8];
        v8bf blo = *(const v8bf*)bp;
        v8bf bhi = *(const v8bf*)(bp + 16);
        v16bf b;
#pragma unroll
        for (int i = 0; i < 8; ++i) { b[i] = blo[i]; b[i + 8] = bhi[i]; }
        acc = __builtin_amdgcn_wmma_f32_16x16x32_bf16(
            false, a[kt], false, b, (short)0, acc, false, false);
      }
      // acc[r] = h[M = r+8*hi][N = n*16+lane16]; fold the h@W2 GEMV in place
      float bn = sB1[n * 16 + lane16];
      float w2 = sW2[n * 16 + lane16];
#pragma unroll
      for (int r = 0; r < 8; ++r)
        part[r] += fast_tanh(acc[r] + bn) * w2;
    }

    // reduce over the 16 lanes (N values) inside each half-wave
#pragma unroll
    for (int m = 1; m < 16; m <<= 1)
#pragma unroll
      for (int r = 0; r < 8; ++r)
        part[r] += __shfl_xor(part[r], m, 32);

    float myPart = part[0];
#pragma unroll
    for (int r = 1; r < 8; ++r)
      if (lane16 == r) myPart = part[r];

    if (lane16 < 8) {
      int node = tile * 16 + hi * 8 + lane16;   // row M = lane16 + 8*hi
      float ev = __expf(myPart + b2v);
      eArr[node] = ev;
      atomicAdd(&denom[batch[node]], ev);
    }
  }
}

// ---------------- Pass 1b: normalize in place: eArr[n] /= (denom[batch[n]] + 1e-8)
__global__ __launch_bounds__(256) void ap_norm(
    const int* __restrict__ batch, const float* __restrict__ denom,
    float* __restrict__ eArr, int nNodes)
{
  int n = blockIdx.x * blockDim.x + threadIdx.x;
  if (n < nNodes)
    eArr[n] = eArr[n] / (denom[batch[n]] + 1e-8f);
}

// ---------------- Pass 2: out[g] += x[n] * w[n], batch sorted -> local acc + boundary flush
__global__ __launch_bounds__(256) void ap_pass2(
    const float* __restrict__ x, const int* __restrict__ batch,
    const float* __restrict__ wArr, float* __restrict__ out, int nNodes)
{
  int col = threadIdx.x;                    // one hidden column per thread
  int start = blockIdx.x * CHUNK;
  if (start >= nNodes) return;
  int end = min(start + CHUNK, nNodes);

  int curg = batch[start];
  float acc = 0.f;
  for (int node = start; node < end; ++node) {
    int g = batch[node];                    // uniform -> scalarized load
    if (g != curg) {
      atomicAdd(&out[(size_t)curg * HIDDEN + col], acc);
      acc = 0.f;
      curg = g;
    }
    acc = fmaf(x[(size_t)node * HIDDEN + col], wArr[node], acc);
  }
  atomicAdd(&out[(size_t)curg * HIDDEN + col], acc);
}

extern "C" void kernel_launch(void* const* d_in, const int* in_sizes, int n_in,
                              void* d_out, int out_size, void* d_ws, size_t ws_size,
                              hipStream_t stream) {
  const float* x     = (const float*)d_in[0];
  const int*   batch = (const int*)d_in[1];
  const float* W1    = (const float*)d_in[2];
  const float* b1    = (const float*)d_in[3];
  const float* W2    = (const float*)d_in[4];
  const float* b2    = (const float*)d_in[5];
  float* out = (float*)d_out;

  const int nNodes  = in_sizes[0] / HIDDEN;   // 500000
  const int nGraphs = out_size / HIDDEN;      // 8192
  const int nTiles  = nNodes / 16;            // 31250 (exact)

  float* eArr  = (float*)d_ws;                // [nNodes]  (e, then normalized weights)
  float* denom = eArr + nNodes;               // [nGraphs]

  hipMemsetAsync(out, 0, (size_t)out_size * sizeof(float), stream);
  hipMemsetAsync(denom, 0, (size_t)nGraphs * sizeof(float), stream);

  ap_pass1<<<1024, 256, 0, stream>>>(x, batch, W1, b1, W2, b2, eArr, denom, nTiles);

  int blocksN = (nNodes + 255) / 256;
  ap_norm<<<blocksN, 256, 0, stream>>>(batch, denom, eArr, nNodes);

  int blocks2 = (nNodes + CHUNK - 1) / CHUNK;
  ap_pass2<<<blocks2, 256, 0, stream>>>(x, batch, eArr, out, nNodes);
}
